// VectorQuantizer_5798205849734
// MI455X (gfx1250) — matrix operand order repaired and checked by reading, not compiled, not run on hardware
//
#include <hip/hip_runtime.h>
#include <cstdint>

#define N_TOK 32768
#define DIM 512
#define KCODES 4096
#define NCHUNK 16            // DIM / 32 k-steps per tile
#define WAVES_PER_BLOCK 8
#define GRID_MAIN 256        // N_TOK / (WAVES_PER_BLOCK * 16)

typedef __bf16 bf16_t;
typedef __attribute__((ext_vector_type(16))) __bf16 v16bf;
typedef __attribute__((ext_vector_type(8)))  float  v8f;

// round-to-nearest-even fp32 -> bf16
__device__ __forceinline__ bf16_t f2bf(float x) {
  union { float f; uint32_t u; } v; v.f = x;
  uint32_t r = v.u + 0x7FFFu + ((v.u >> 16) & 1u);
  union { uint16_t s; bf16_t b; } o; o.s = (uint16_t)(r >> 16);
  return o.b;
}

// ---------------------------------------------------------------------------
// Kernel 1: convert codebook + gate_w to bf16 in workspace, compute per-code
// squared norms, zero the loss accumulator.
// ---------------------------------------------------------------------------
__global__ __launch_bounds__(128) void vq_prep(const float* __restrict__ codebook,
                                               const float* __restrict__ gate_w,
                                               bf16_t* __restrict__ cb_bf,
                                               bf16_t* __restrict__ gw_bf,
                                               float* __restrict__ cnorm,
                                               float* __restrict__ loss_acc) {
  int r = blockIdx.x;
  int tid = threadIdx.x;
  if (r == 0 && tid == 0) *loss_acc = 0.0f;
  if (r < KCODES) {
    const float* src = codebook + (size_t)r * DIM;
    bf16_t* dst = cb_bf + (size_t)r * DIM;
    float s = 0.0f;
    for (int i = tid; i < DIM; i += 128) {
      float x = src[i];
      dst[i] = f2bf(x);
      s += x * x;
    }
    for (int off = 16; off >= 1; off >>= 1) s += __shfl_xor(s, off, 32);
    __shared__ float red[4];
    if ((tid & 31) == 0) red[tid >> 5] = s;
    __syncthreads();
    if (tid == 0) cnorm[r] = red[0] + red[1] + red[2] + red[3];
  } else {
    int g = r - KCODES;
    const float* src = gate_w + (size_t)g * DIM;
    bf16_t* dst = gw_bf + (size_t)g * DIM;
    for (int i = tid; i < DIM; i += 128) dst[i] = f2bf(src[i]);
  }
}

// ---------------------------------------------------------------------------
// A-fragment loader: 16 tokens x 512 dims held in registers as 16 v16bf frags.
// A k order used here (contiguous, B matches): lane l<16 -> row l, K=0..15; lane l+16 ->
// row l, K=16..31.
// ---------------------------------------------------------------------------
__device__ __forceinline__ void load_a_frags(const float* __restrict__ inp,
                                             int row_base, int lane,
                                             v16bf a[NCHUNK]) {
  int mrow = lane & 15;
  int koff = (lane >> 4) * 16;
  const float* arow = inp + (size_t)(row_base + mrow) * DIM;
#pragma unroll
  for (int c = 0; c < NCHUNK; ++c) {
    int d0 = c * 32 + koff;
    float tmp[16];
    *(float4*)(tmp + 0)  = *(const float4*)(arow + d0);
    *(float4*)(tmp + 4)  = *(const float4*)(arow + d0 + 4);
    *(float4*)(tmp + 8)  = *(const float4*)(arow + d0 + 8);
    *(float4*)(tmp + 12) = *(const float4*)(arow + d0 + 12);
    v16bf t;
#pragma unroll
    for (int j = 0; j < 16; ++j) t[j] = f2bf(tmp[j]);
    a[c] = t;
  }
}

// ---------------------------------------------------------------------------
// Global -> register / register -> LDS halves of the tile staging, split so
// the global loads of tile ct+1 can be issued before the WMMAs of tile ct
// (double-buffered LDS).  Fragment-order LDS layout: chunk ch (kc=ch>>5,
// l=ch&31) holds 16 contiguous bf16 (32B) so per-kc fragment reads are one
// contiguous 1KB span per wave (conflict-free ds_load_b128).
// ---------------------------------------------------------------------------
__device__ __forceinline__ void stage_load(const bf16_t* __restrict__ mat,
                                           int row0, int tid, uint4 r[4]) {
#pragma unroll
  for (int i = 0; i < 2; ++i) {
    int ch = tid + i * 256;
    int kc = ch >> 5, l = ch & 31;
    const uint4* src = (const uint4*)(mat + (size_t)(row0 + (l & 15)) * DIM +
                                      kc * 32 + (l >> 4) * 16);
    r[2 * i]     = src[0];
    r[2 * i + 1] = src[1];
  }
}

__device__ __forceinline__ void stage_store(bf16_t* sB, int tid,
                                            const uint4 r[4]) {
#pragma unroll
  for (int i = 0; i < 2; ++i) {
    int ch = tid + i * 256;
    uint4* dst = (uint4*)((char*)sB + ch * 32);
    dst[0] = r[2 * i];
    dst[1] = r[2 * i + 1];
  }
}

// 16-step WMMA accumulation over one staged tile with register double-buffered
// B fragments (next ds_load issued before current wmma).
__device__ __forceinline__ v8f tile_gemm(const v16bf a[NCHUNK],
                                         const bf16_t* __restrict__ cur,
                                         int lane) {
  v8f acc = {};
  v16bf b = *(const v16bf*)((const char*)cur + lane * 32);
#pragma unroll
  for (int kc = 0; kc < NCHUNK; ++kc) {
    v16bf bn = b;
    if (kc < NCHUNK - 1)
      bn = *(const v16bf*)((const char*)cur + (kc + 1) * 1024 + lane * 32);
    acc = __builtin_amdgcn_wmma_f32_16x16x32_bf16(false, a[kc], false, b,
                                                  (short)0, acc, false, false);
    b = bn;
  }
  return acc;
}

// ---------------------------------------------------------------------------
// Kernel 2: fused distance GEMM + argmin. Each wave owns 16 tokens, keeps A
// in registers, loops over 256 code tiles double-buffered through LDS.
// dist = ||c||^2 - 2 x.c (row norm constant w.r.t. argmin).
// ---------------------------------------------------------------------------
__global__ __launch_bounds__(256, 1) void vq_argmin_kernel(
    const float* __restrict__ inp, const bf16_t* __restrict__ cb_bf,
    const float* __restrict__ cnorm, int* __restrict__ idx_out) {
  __shared__ __align__(32) bf16_t sB[2][16 * DIM];   // 2 x 16 KB
  int tid = threadIdx.x;
  int wave = tid >> 5, lane = tid & 31;
  int row_base = (blockIdx.x * WAVES_PER_BLOCK + wave) * 16;
  int mrow = lane & 15, half = lane >> 4;

  v16bf a[NCHUNK];
  load_a_frags(inp, row_base, lane, a);

  float best[8];
  int bidx[8];
#pragma unroll
  for (int v = 0; v < 8; ++v) { best[v] = 3.4e38f; bidx[v] = 0; }

  uint4 pre[4];
  stage_load(cb_bf, 0, tid, pre);
  stage_store(sB[0], tid, pre);
  __syncthreads();

  const int NT = KCODES / 16;
  for (int ct = 0; ct < NT; ++ct) {
    const bf16_t* cur = sB[ct & 1];
    if (ct + 1 < NT) stage_load(cb_bf, (ct + 1) * 16, tid, pre);  // prefetch

    float cn = cnorm[ct * 16 + mrow];
    v8f acc = tile_gemm(a, cur, lane);

    int code = ct * 16 + mrow;
#pragma unroll
    for (int v = 0; v < 8; ++v) {
      float dist = cn - 2.0f * acc[v];
      if (dist < best[v]) { best[v] = dist; bidx[v] = code; }
    }

    if (ct + 1 < NT) stage_store(sB[(ct + 1) & 1], tid, pre);
    __syncthreads();
  }

  // C layout: VGPR v -> row v (lanes 0-15) / row v+8 (lanes 16-31), col=lane%16
#pragma unroll
  for (int v = 0; v < 8; ++v) {
    float bv = best[v];
    int bi = bidx[v];
    for (int off = 8; off >= 1; off >>= 1) {
      float ov = __shfl_xor(bv, off, 32);
      int oi = __shfl_xor(bi, off, 32);
      if (ov < bv || (ov == bv && oi < bi)) { bv = ov; bi = oi; }
    }
    if (mrow == 0) idx_out[row_base + half * 8 + v] = bi;
  }
}

// ---------------------------------------------------------------------------
// Kernel 3: gate GEMM (sigmoid(x @ gate_w^T + b)) fused with gather of the
// selected codes, output = x + q*gate, and MSE-loss partial accumulation.
// ---------------------------------------------------------------------------
__global__ __launch_bounds__(256, 1) void vq_gate_out_kernel(
    const float* __restrict__ inp, const float* __restrict__ codebook,
    const bf16_t* __restrict__ gw_bf, const float* __restrict__ gate_b,
    const int* __restrict__ idx, float* __restrict__ out,
    float* __restrict__ loss_acc) {
  __shared__ __align__(32) bf16_t sB[2][16 * DIM];
  int tid = threadIdx.x;
  int wave = tid >> 5, lane = tid & 31;
  int row_base = (blockIdx.x * WAVES_PER_BLOCK + wave) * 16;
  int mrow = lane & 15, half = lane >> 4;

  v16bf a[NCHUNK];
  load_a_frags(inp, row_base, lane, a);

  int qrow[8];
#pragma unroll
  for (int v = 0; v < 8; ++v) qrow[v] = idx[row_base + half * 8 + v];

  uint4 pre[4];
  stage_load(gw_bf, 0, tid, pre);
  stage_store(sB[0], tid, pre);
  __syncthreads();

  float lsum = 0.0f;
  const int NT = DIM / 16;
  for (int ct = 0; ct < NT; ++ct) {
    const bf16_t* cur = sB[ct & 1];
    if (ct + 1 < NT) stage_load(gw_bf, (ct + 1) * 16, tid, pre);  // prefetch

    float bias = gate_b[ct * 16 + mrow];
    v8f acc = tile_gemm(a, cur, lane);

    int col = ct * 16 + mrow;
#pragma unroll
    for (int v = 0; v < 8; ++v) {
      int row = row_base + half * 8 + v;
      float g = 1.0f / (1.0f + __expf(-(acc[v] + bias)));
      float x = inp[(size_t)row * DIM + col];
      float q = codebook[(size_t)qrow[v] * DIM + col];
      out[(size_t)row * DIM + col] = x + q * g;
      float d = q - x;
      lsum += d * d;
    }

    if (ct + 1 < NT) stage_store(sB[(ct + 1) & 1], tid, pre);
    __syncthreads();
  }

  for (int off = 16; off >= 1; off >>= 1) lsum += __shfl_xor(lsum, off, 32);
  if (lane == 0) atomicAdd(loss_acc, lsum);
}

// ---------------------------------------------------------------------------
// Kernel 4: loss = (1 + 0.25) * mean((q - x)^2)
// ---------------------------------------------------------------------------
__global__ void vq_finalize(const float* __restrict__ loss_acc,
                            float* __restrict__ out_loss) {
  *out_loss = 1.25f * (*loss_acc) / (float)(N_TOK * DIM);
}

extern "C" void kernel_launch(void* const* d_in, const int* in_sizes, int n_in,
                              void* d_out, int out_size, void* d_ws, size_t ws_size,
                              hipStream_t stream) {
  const float* inputs   = (const float*)d_in[0];  // [8,4096,512]
  const float* codebook = (const float*)d_in[1];  // [4096,512]
  const float* gate_w   = (const float*)d_in[2];  // [512,512]
  const float* gate_b   = (const float*)d_in[3];  // [512]
  float* out = (float*)d_out;                     // [N*D] output + [1] loss

  char* ws = (char*)d_ws;
  size_t off = 0;
  bf16_t* cb_bf = (bf16_t*)(ws + off); off += (size_t)KCODES * DIM * 2;  // 4 MB
  bf16_t* gw_bf = (bf16_t*)(ws + off); off += (size_t)DIM * DIM * 2;     // 512 KB
  float* cnorm  = (float*)(ws + off);  off += (size_t)KCODES * 4;        // 16 KB
  int*   idx    = (int*)(ws + off);    off += (size_t)N_TOK * 4;         // 128 KB
  float* loss_acc = (float*)(ws + off);

  vq_prep<<<KCODES + DIM, 128, 0, stream>>>(codebook, gate_w, cb_bf, gw_bf,
                                            cnorm, loss_acc);
  vq_argmin_kernel<<<GRID_MAIN, 256, 0, stream>>>(inputs, cb_bf, cnorm, idx);
  vq_gate_out_kernel<<<GRID_MAIN, 256, 0, stream>>>(inputs, codebook, gw_bf,
                                                    gate_b, idx, out, loss_acc);
  vq_finalize<<<1, 1, 0, stream>>>(loss_acc, out + ((size_t)out_size - 1));
}
